// QuantLinearHQQ_78881369358652
// MI455X (gfx1250) — matrix operand order repaired and checked by reading, not compiled, not run on hardware
//
#include <hip/hip_runtime.h>

typedef __attribute__((ext_vector_type(16))) _Float16 v16h;
typedef __attribute__((ext_vector_type(8)))  float    v8f;
typedef __attribute__((ext_vector_type(2)))  _Float16 h2;
typedef __attribute__((ext_vector_type(4)))  unsigned u32x4;
typedef __attribute__((ext_vector_type(8)))  int      i32x8;
typedef __attribute__((ext_vector_type(4)))  int      i32x4;

#define K_TOT 4096
#define N_TOT 4096

#define BM 128
#define BN 128
#define BK 32
#define LDA 40   // 32 + 8 half padding (80B row stride) == TDM pad 16DW + 4DW
#define LDB 40

union FragU { uint4 u4[2]; v16h h; };
union H2U   { unsigned u; h2 h; };

// Dequant 8 nibbles of one qweight int32 into 8 ordered fp16 values.
// fp16 magic: 0x6400|nib == fp16(1024+nib); value = (1024+w)*s + (-s*(1024+z)).
__device__ __forceinline__ uint4 dequant8(unsigned q, h2 s2, h2 c2) {
    unsigned p04 = ( q         & 0x000F000Fu) | 0x64006400u;
    unsigned p15 = ((q >>  4)  & 0x000F000Fu) | 0x64006400u;
    unsigned p26 = ((q >>  8)  & 0x000F000Fu) | 0x64006400u;
    unsigned p37 = ((q >> 12)  & 0x000F000Fu) | 0x64006400u;
    unsigned k01 = __builtin_amdgcn_perm(p15, p04, 0x05040100u);
    unsigned k23 = __builtin_amdgcn_perm(p37, p26, 0x05040100u);
    unsigned k45 = __builtin_amdgcn_perm(p15, p04, 0x07060302u);
    unsigned k67 = __builtin_amdgcn_perm(p37, p26, 0x07060302u);
    H2U a, b, c, d;
    a.u = k01; b.u = k23; c.u = k45; d.u = k67;
    a.h = a.h * s2 + c2;   // v_pk_fma_f16
    b.h = b.h * s2 + c2;
    c.h = c.h * s2 + c2;
    d.h = d.h * s2 + c2;
    return make_uint4(a.u, b.u, c.u, d.u);
}

__launch_bounds__(256)
__global__ void w4a16_wmma_gemm(const _Float16* __restrict__ X,
                                const int*      __restrict__ QW,
                                const _Float16* __restrict__ SC,
                                const _Float16* __restrict__ QZ,
                                const _Float16* __restrict__ BIAS,
                                _Float16*       __restrict__ OUT,
                                int Mrows)
{
    __shared__ _Float16 As[2][BM * LDA];
    __shared__ _Float16 Bs[2][BN * LDB];

    const int tid  = threadIdx.x;
    const int lane = tid & 31;
    const int wave = tid >> 5;
    const int wm   = wave & 3;          // 4 waves along M
    const int wn   = wave >> 2;         // 2 waves along N
    const int m0   = blockIdx.y * BM;
    const int n0   = blockIdx.x * BN;

    // ---- B dequant mapping: 1 column per thread (mod), 2 qweight rows ----
    const int bcol = tid & 127;         // column within tile
    const int brp  = tid >> 7;          // 0..1 -> rows {0,1} or {2,3} of 4
    const int ncol = n0 + bcol;         // global out column

    v8f acc[2][4] = {};

    // per-group packed dequant constants (group = 64 K values = 2 K-tiles)
    h2 s2 = {};
    h2 c2 = {};
    auto load_group = [&](int g) {
        float sF = (float)SC[(size_t)g * N_TOT + ncol];
        float zF = (float)QZ[(size_t)g * N_TOT + ncol];
        _Float16 sh = (_Float16)sF;
        _Float16 ch = (_Float16)(-sF * (1024.0f + zF));
        h2 st = {sh, sh};
        h2 ct = {ch, ch};
        s2 = st;
        c2 = ct;
    };

    // ---- A tile via Tensor Data Mover: one 2D descriptor per stage ----
    // Tile = 32 x 128 fp16 of row-major X, TDM pad recreates the 80B LDS row
    // stride (pad_interval=3 -> every 16 DWORDs, pad_amount=3 -> +4 DWORDs).
    const unsigned ldsA[2] = { (unsigned)(size_t)(void*)&As[0][0],
                               (unsigned)(size_t)(void*)&As[1][0] };
    const unsigned long long gA0 =
        (unsigned long long)(size_t)(X + (size_t)m0 * K_TOT);

    auto tdm_load_a = [&](int kt, int buf) {
        const unsigned long long ga = gA0 + (unsigned long long)kt * (BK * 2);
        u32x4 g0;
        g0[0] = 1u;                                   // count=1 (valid, user)
        g0[1] = ldsA[buf];                            // lds_addr (bytes)
        g0[2] = (unsigned)ga;                         // global_addr[31:0]
        g0[3] = ((unsigned)(ga >> 32) & 0x01FFFFFFu)  // global_addr[56:32]
                | 0x80000000u;                        // type=2 ("image")
        i32x8 g1;
        g1[0] = (int)((1u << 16)      // data_size = 2 bytes
                    | (1u << 20)      // pad_enable
                    | (3u << 22)      // pad_interval: 16 DWORDs
                    | (3u << 25));    // pad_amount:   4 DWORDs
        g1[1] = (int)(((unsigned)K_TOT & 0xFFFFu) << 16);            // dim0 lo
        g1[2] = (int)((((unsigned)K_TOT >> 16) & 0xFFFFu)
                    | (((unsigned)Mrows & 0xFFFFu) << 16));          // dim0 hi | dim1 lo
        g1[3] = (int)((((unsigned)Mrows >> 16) & 0xFFFFu)
                    | (32u << 16));                                  // dim1 hi | tile_dim0
        g1[4] = 128;                                                 // tile_dim1 (tile_dim2=0)
        g1[5] = K_TOT;                                               // dim0_stride lo32
        g1[6] = 0;
        g1[7] = 0;
        i32x4 gz4 = {0, 0, 0, 0};
        i32x8 gz8 = {0, 0, 0, 0, 0, 0, 0, 0};
        __builtin_amdgcn_tensor_load_to_lds(g0, g1, gz4, gz4, gz8, 0);
    };

    auto load_stage_b = [&](int kt, int buf) {
        const int k0 = kt * BK;
        const int qrow = (k0 >> 3) + brp * 2;    // int32 row covers 8 K values
        #pragma unroll
        for (int r = 0; r < 2; ++r) {
            unsigned q = (unsigned)QW[(size_t)(qrow + r) * N_TOT + ncol];
            uint4 w = dequant8(q, s2, c2);
            *(uint4*)&Bs[buf][bcol * LDB + (brp * 2 + r) * 8] = w;
        }
    };

    auto compute_stage = [&](int buf) {
        const int mrow = lane & 15;
        const int hh   = lane >> 4;
        v16h afrag[2], bfrag[4];
        // A frag: ISA 16-bit A 16x32 layout. v0..v3: K[h*8..+7], v4..v7: K[16+h*8..+7]
        #pragma unroll
        for (int mt = 0; mt < 2; ++mt) {
            FragU f;
            const int base = (wm * 32 + mt * 16 + mrow) * LDA + hh * 8;
            f.u4[0] = *(const uint4*)&As[buf][base];
            f.u4[1] = *(const uint4*)&As[buf][base + 16];
            afrag[mt] = f.h;
        }
        // B frag: same K order as A: v0..v3: K[h*8..+7], v4..v7: K[16+h*8..+7], column = lane&15
        #pragma unroll
        for (int nt = 0; nt < 4; ++nt) {
            FragU f;
            const int base = (wn * 64 + nt * 16 + mrow) * LDB + hh * 8;
            f.u4[0] = *(const uint4*)&Bs[buf][base];
            f.u4[1] = *(const uint4*)&Bs[buf][base + 16];
            bfrag[nt] = f.h;
        }
        #pragma unroll
        for (int mt = 0; mt < 2; ++mt)
            #pragma unroll
            for (int nt = 0; nt < 4; ++nt)
                acc[mt][nt] = __builtin_amdgcn_wmma_f32_16x16x32_f16(
                    false, afrag[mt], false, bfrag[nt],
                    (short)0, acc[mt][nt], false, false);
    };

    const int NT = K_TOT / BK;   // 128
    load_group(0);
    if (wave == 0) tdm_load_a(0, 0);      // TDM ignores EXEC: single-wave issue
    load_stage_b(0, 0);
    if (wave == 0) __builtin_amdgcn_s_wait_tensorcnt(0);
    __syncthreads();

    for (int kt = 0; kt < NT; ++kt) {
        const int cur = kt & 1;
        if (kt + 1 < NT) {
            if (kt + 2 < NT)              // pull qweight k+2 toward GL2
                __builtin_prefetch(
                    &QW[(size_t)(((kt + 2) * BK) >> 3) * N_TOT + ncol], 0, 1);
            if (((kt + 1) & 1) == 0)      // uniform branch: new quant group
                load_group((kt + 1) >> 1);
            if (wave == 0) tdm_load_a(kt + 1, cur ^ 1);
            load_stage_b(kt + 1, cur ^ 1);
        }
        compute_stage(cur);
        if (wave == 0) __builtin_amdgcn_s_wait_tensorcnt(0);  // drain TDM
        __syncthreads();
    }

    // ---- epilogue: C/D layout -> row = (lane>>4)*8 + r, col = lane&15 ----
    #pragma unroll
    for (int nt = 0; nt < 4; ++nt) {
        const int   cg = n0 + wn * 64 + nt * 16 + (lane & 15);
        const float bv = (float)BIAS[cg];
        #pragma unroll
        for (int mt = 0; mt < 2; ++mt) {
            const int rbase = m0 + wm * 32 + mt * 16 + (lane >> 4) * 8;
            #pragma unroll
            for (int r = 0; r < 8; ++r) {
                OUT[(size_t)(rbase + r) * N_TOT + cg] =
                    (_Float16)(acc[mt][nt][r] + bv);
            }
        }
    }
}

extern "C" void kernel_launch(void* const* d_in, const int* in_sizes, int n_in,
                              void* d_out, int out_size, void* d_ws, size_t ws_size,
                              hipStream_t stream) {
    const _Float16* x    = (const _Float16*)d_in[0];
    const int*      qw   = (const int*)d_in[1];
    const _Float16* sc   = (const _Float16*)d_in[2];
    const _Float16* qz   = (const _Float16*)d_in[3];
    const _Float16* bias = (const _Float16*)d_in[4];
    _Float16*       out  = (_Float16*)d_out;

    const int m_tot = in_sizes[0] / K_TOT;   // B*S = 8192
    dim3 grid(N_TOT / BN, m_tot / BM);       // 32 x 64
    w4a16_wmma_gemm<<<grid, 256, 0, stream>>>(x, qw, sc, qz, bias, out, m_tot);
}
